// Triton_17377437680121
// MI455X (gfx1250) — matrix operand order repaired and checked by reading, not compiled, not run on hardware
//
#include <hip/hip_runtime.h>
#include <hip/hip_bf16.h>

typedef _Float16 v16h __attribute__((ext_vector_type(16)));
typedef _Float16 v8h  __attribute__((ext_vector_type(8)));
typedef float    v8f  __attribute__((ext_vector_type(8)));
typedef float    v4f  __attribute__((ext_vector_type(4)));

namespace {
constexpr int DI     = 1024;   // K
constexpr int DO     = 1024;   // N
constexpr int KT     = 32;     // K per WMMA step
constexpr int KTP    = KT + 8; // padded LDS stride (halves) to avoid bank conflicts
constexpr int WN     = 64;     // N columns per workgroup (4 WMMA subtiles)
constexpr int WM     = 16;     // token rows per wave
constexpr int WAVES  = 8;      // wave32 x 8 = 256 threads
constexpr int BM     = WM * WAVES;  // 128 token rows per block
constexpr int THREADS = 32 * WAVES;
}

// -------- pre-pass: Wv/Wt f32 -> f16 (one-time, L2-resident 8MB -> 4MB) ------
__global__ __launch_bounds__(256) void convert_w_f16(
    const float* __restrict__ w, _Float16* __restrict__ h, int n8)
{
    const int i = blockIdx.x * 256 + threadIdx.x;
    if (i >= n8) return;
    const v4f a = *(const v4f*)(w + (long)i * 8);
    const v4f b = *(const v4f*)(w + (long)i * 8 + 4);
    v8h o;
#pragma unroll
    for (int j = 0; j < 4; ++j) {
        o[j]     = (_Float16)a[j];
        o[j + 4] = (_Float16)b[j];
    }
    *(v8h*)(h + (long)i * 8) = o;
}

// -------- main: y[m,n] = sum_k x[m,k] * (mask[m] ? Wv : Wt)[n,k] -------------
// gated dual-GEMM: A_v = mask?x:0, A_t = mask?0:x,
// D = A_v x Wv^T + A_t x Wt^T  (f16 WMMA, f32 accumulate)
__global__ __launch_bounds__(THREADS) void moe2_wmma_f16(
    const float*    __restrict__ x,
    const int*      __restrict__ vmask,
    const _Float16* __restrict__ hWv,
    const _Float16* __restrict__ hWt,
    float*          __restrict__ y)
{
    __shared__ __align__(16) _Float16 lwv[WN * KTP];
    __shared__ __align__(16) _Float16 lwt[WN * KTP];

    const int tid  = threadIdx.x;
    const int wave = tid >> 5;
    const int lane = tid & 31;
    const int lrow = lane & 15;   // A-row / B-col within tile
    const int hi   = lane >> 4;   // lane half selects K group

    const int mBase = blockIdx.x * BM + wave * WM;
    const int nBase = blockIdx.y * WN;

    const long arow = (long)mBase + lrow;         // global token row
    const bool mrow = vmask[arow] != 0;           // expert select for this row

    v8f accv[4] = {};
    v8f acct[4] = {};

    // --- async W staging: 256 threads cover 64 n-rows x 32 k, 8 halves (16B) each
    const int fn = tid >> 2;            // 0..63  (n within tile)
    const int fk = (tid & 3) * 8;       // 0,8,16,24 (k chunk, halves)
    const unsigned ldsV = (unsigned)(size_t)&lwv[fn * KTP + fk];
    const unsigned ldsT = (unsigned)(size_t)&lwt[fn * KTP + fk];
    int voff = ((nBase + fn) * DI + fk) * (int)sizeof(_Float16); // byte offset, same for both experts

    // --- A fragment addressing: lane holds K = 16*hi .. 16*hi+15 contiguous (matches B)
    const float* xp = x + arow * DI + (hi ? 16 : 0);
    const int bklo  = hi ? 16 : 0;      // B fragment: K = bklo..bklo+15 contiguous

    for (int k0 = 0; k0 < DI; k0 += KT) {
        __syncthreads();  // all waves done reading LDS slab from previous step

        // issue async LDS fills (ASYNCcnt); overlap with A load/convert below
        asm volatile(
            "global_load_async_to_lds_b128 %0, %2, %3 offset:0\n\t"
            "global_load_async_to_lds_b128 %1, %2, %4 offset:0"
            :
            : "v"(ldsV), "v"(ldsT), "v"(voff), "s"(hWv), "s"(hWt)
            : "memory");
        voff += KT * (int)sizeof(_Float16);

        // load + convert A fragment (one contiguous 16-float run per lane)
        v16h af, zf = {};
        {
            v4f a0 = *(const v4f*)(xp + k0);
            v4f a1 = *(const v4f*)(xp + k0 + 4);
            v4f a2 = *(const v4f*)(xp + k0 + 8);
            v4f a3 = *(const v4f*)(xp + k0 + 12);
#pragma unroll
            for (int i = 0; i < 4; ++i) {
                af[i]      = (_Float16)a0[i];
                af[i + 4]  = (_Float16)a1[i];
                af[i + 8]  = (_Float16)a2[i];
                af[i + 12] = (_Float16)a3[i];
            }
        }
        const v16h av = mrow ? af : zf;   // visual expert rows
        const v16h at = mrow ? zf : af;   // text expert rows

        // publish the async slab to the whole workgroup
        asm volatile("s_wait_asynccnt 0" ::: "memory");
        __syncthreads();

#pragma unroll
        for (int t = 0; t < 4; ++t) {
            const v16h bv = *(const v16h*)&lwv[(t * 16 + lrow) * KTP + bklo];
            const v16h bt = *(const v16h*)&lwt[(t * 16 + lrow) * KTP + bklo];
            accv[t] = __builtin_amdgcn_wmma_f32_16x16x32_f16(
                false, av, false, bv, (short)0, accv[t], false, false);
            acct[t] = __builtin_amdgcn_wmma_f32_16x16x32_f16(
                false, at, false, bt, (short)0, acct[t], false, false);
        }
    }

    // epilogue: D layout -> VGPR r holds row (r + 8*hi), lane%16 is the column
#pragma unroll
    for (int t = 0; t < 4; ++t) {
        v8f d = accv[t] + acct[t];
        const long col = (long)nBase + t * 16 + lrow;
#pragma unroll
        for (int r = 0; r < 8; ++r) {
            const long m = (long)mBase + r + 8 * hi;
            y[m * DO + col] = d[r];
        }
    }
}

extern "C" void kernel_launch(void* const* d_in, const int* in_sizes, int n_in,
                              void* d_out, int out_size, void* d_ws, size_t ws_size,
                              hipStream_t stream) {
    const float* x  = (const float*)d_in[0];
    const int*   vm = (const int*)d_in[1];
    const float* Wv = (const float*)d_in[2];
    const float* Wt = (const float*)d_in[3];
    float*       y  = (float*)d_out;

    _Float16* hv = (_Float16*)d_ws;                 // 2 MB
    _Float16* ht = hv + (size_t)DO * DI;            // 2 MB

    const int n8 = DO * DI / 8;                     // 131072 vector-8 chunks
    convert_w_f16<<<n8 / 256, 256, 0, stream>>>(Wv, hv, n8);
    convert_w_f16<<<n8 / 256, 256, 0, stream>>>(Wt, ht, n8);

    const int M = in_sizes[1];                      // B*S tokens (mask element count)
    dim3 grid(M / BM, DO / WN);
    moe2_wmma_f16<<<grid, THREADS, 0, stream>>>(x, vm, hv, ht, y);
}
